// MultiShapeNet0_20005957665408
// MI455X (gfx1250) — matrix operand order repaired and checked by reading, not compiled, not run on hardware
//
#include <hip/hip_runtime.h>

typedef float    f4   __attribute__((ext_vector_type(4)));
typedef float    v8f  __attribute__((ext_vector_type(8)));
typedef _Float16 v8h  __attribute__((ext_vector_type(8)));
typedef _Float16 v16h __attribute__((ext_vector_type(16)));

#define HID   128
#define NBLK  8
#define NLAY  2
#define W0f   1.0f
#define SWAPX16_IMM 0x401F   // group-of-32: xor=0x10, or=0, and=0x1f

// ---------------------------------------------------------------------------
// Pre-pass: W_blocks f32 [16][128k][128n]  ->  Wt f16 [16][128n][128k]
// (out-feature major, so WMMA A-fragments read contiguous K)
// ---------------------------------------------------------------------------
__global__ __launch_bounds__(256) void wt_f16_kernel(const float* __restrict__ W,
                                                     _Float16* __restrict__ Wt) {
  int idx = blockIdx.x * 256 + threadIdx.x;      // 16*128*128 threads
  int k = idx & 127;
  int n = (idx >> 7) & 127;
  int m = idx >> 14;                             // block*2 + layer, 0..15
  Wt[(m * 128 + n) * 128 + k] = (_Float16)W[(m * 128 + k) * 128 + n];
}

__device__ __forceinline__ float swap16(float v) {
  return __int_as_float(__builtin_amdgcn_ds_swizzle(__float_as_int(v), SWAPX16_IMM));
}

// Build the 4 B-fragments (K=in-features x N=points, f16) from 64 f32 values
// held in h-layout (feature 16g+8*sub+e at h[g*8+e]).  Lane-local (swap16 results unused).
__device__ __forceinline__ void build_b(const float* __restrict__ src, bool subsel,
                                        v16h b[4]) {
#pragma unroll
  for (int kt = 0; kt < 4; ++kt) {
#pragma unroll
    for (int e = 0; e < 8; ++e) {
      const float v0 = src[(2 * kt) * 8 + e];      // own feature 32kt + 8*sub + e
      const float v1 = src[(2 * kt + 1) * 8 + e];  // own feature 32kt + 16 + 8*sub + e
      const float w0 = swap16(v0);                 // partner half-wave's v0
      const float w1 = swap16(v1);                 // partner half-wave's v1
      const float lo = v0;                         // feature 32kt + 8*sub + e
      const float hi = v1;                         // feature 32kt + 16 + 8*sub + e
      b[kt][e]     = (_Float16)lo;
      b[kt][8 + e] = (_Float16)hi;
    }
  }
}

// ---------------------------------------------------------------------------
// Main kernel (transposed GEMMs): D = Wt_tile (A, MxK) x h^T (B, KxN).
// D layout == h register layout, so FiLM / bias+sin / residual are all
// lane-local; h -> B-fragment needs no exchange either.
// ---------------------------------------------------------------------------
__global__ __launch_bounds__(256) void siren_film_wmma(
    const float* __restrict__ x,        // [N,4]
    const float* __restrict__ gammas,   // [N,8,128]
    const float* __restrict__ betas,    // [N,8,128]
    const float* __restrict__ Wpre,     // [4,128]
    const float* __restrict__ bpre,     // [128]
    const _Float16* __restrict__ Wt,    // [16][128n][128k] f16 (from pre-pass)
    const float* __restrict__ bblk,     // [16][128]
    float* __restrict__ out)            // [N,128]
{
  __shared__ __attribute__((aligned(16))) _Float16 smem_w[NLAY][HID * HID];  // 64KB

  const int tid  = threadIdx.x;
  const int wave = tid >> 5;
  const int lane = tid & 31;
  const int sub  = lane >> 4;           // half-wave select
  const int lm   = lane & 15;           // point index within the 16-point tile
  const bool subsel = (sub != 0);
  const long row = ((long)blockIdx.x * 8 + wave) * 16 + lm;

  // h[g*8+e] = h(point row, feature 16*g + 8*sub + e)
  float h[64];

  // ---- pre-layer: h = sin(W0 * (x @ Wpre + bpre)) : tiny, plain VALU ----
  const f4 xv = *(const f4*)(x + row * 4);
#pragma unroll
  for (int g = 0; g < 8; ++g) {
    const int c0 = 16 * g + 8 * sub;
    f4 s0 = *(const f4*)(bpre + c0);
    f4 s1 = *(const f4*)(bpre + c0 + 4);
#pragma unroll
    for (int k = 0; k < 4; ++k) {
      const f4 w0v = *(const f4*)(Wpre + k * HID + c0);
      const f4 w1v = *(const f4*)(Wpre + k * HID + c0 + 4);
      s0 += xv[k] * w0v;
      s1 += xv[k] * w1v;
    }
#pragma unroll
    for (int e = 0; e < 4; ++e) {
      h[g * 8 + e]     = __sinf(W0f * s0[e]);
      h[g * 8 + 4 + e] = __sinf(W0f * s1[e]);
    }
  }

  for (int blk = 0; blk < NBLK; ++blk) {
    // ---- stage this block's two 128x128 f16 weight matrices into LDS ------
    __syncthreads();   // all waves done reading previous block's weights
    {
      const _Float16* wsrc = Wt + (size_t)blk * NLAY * HID * HID;   // 64 KB
      const unsigned lds_base = (unsigned)(size_t)(&smem_w[0][0]);  // LDS offset
#pragma unroll
      for (int i = 0; i < 16; ++i) {
        const unsigned off  = (unsigned)tid * 16u + (unsigned)i * 4096u;
        const unsigned ldsa = lds_base + off;
        asm volatile("global_load_async_to_lds_b128 %0, %1, %2"
                     :: "v"(ldsa), "v"(off), "s"(wsrc) : "memory");
      }
    }

    // ---- FiLM: h = gamma*h + beta  (the dominant HBM stream, coalesced) ----
    const float* gp = gammas + (row * NBLK + blk) * HID;
    const float* bp = betas  + (row * NBLK + blk) * HID;
#pragma unroll
    for (int g = 0; g < 8; ++g) {
      const int c0 = 16 * g + 8 * sub;
      const f4 ga0 = *(const f4*)(gp + c0);
      const f4 ga1 = *(const f4*)(gp + c0 + 4);
      const f4 be0 = *(const f4*)(bp + c0);
      const f4 be1 = *(const f4*)(bp + c0 + 4);
#pragma unroll
      for (int e = 0; e < 4; ++e) {
        h[g * 8 + e]     = ga0[e] * h[g * 8 + e]     + be0[e];
        h[g * 8 + 4 + e] = ga1[e] * h[g * 8 + 4 + e] + be1[e];
      }
    }

    // ---- prefetch next block's gamma/beta rows (global_prefetch_b8) ----
    if (blk + 1 < NBLK) {
      const char* gn = (const char*)(gammas + (row * NBLK + blk + 1) * HID);
      const char* bn = (const char*)(betas  + (row * NBLK + blk + 1) * HID);
      __builtin_prefetch(gn + sub * 256,       0, 1);
      __builtin_prefetch(gn + sub * 256 + 128, 0, 1);
      __builtin_prefetch(bn + sub * 256,       0, 1);
      __builtin_prefetch(bn + sub * 256 + 128, 0, 1);
    }

    // ---- B-fragments of r = h (layer-0 input), built while async copy runs ----
    v16h b0[4];
    build_b(h, subsel, b0);

    // wait for the async weight staging (own wave), then workgroup-sync
    asm volatile("s_wait_asynccnt 0x0" ::: "memory");
    __syncthreads();

    // =================== layer 0: r1 = sin(W0t x r^T + b) ===================
    float r1[64];
    {
      const _Float16* wl      = &smem_w[0][0];
      const float*    biasrow = bblk + (blk * NLAY + 0) * HID;
#pragma unroll
      for (int mt = 0; mt < 8; ++mt) {
        // preload all 4 K-tile A-fragments (weights) for this M-tile
        const _Float16* wr = wl + (16 * mt + lm) * HID + 8 * sub;
        v16h wa[4];
#pragma unroll
        for (int kt = 0; kt < 4; ++kt) {
          const v8h lo = *(const v8h*)(wr + 32 * kt);
          const v8h hi = *(const v8h*)(wr + 32 * kt + 16);
#pragma unroll
          for (int i = 0; i < 8; ++i) { wa[kt][i] = lo[i]; wa[kt][8 + i] = hi[i]; }
        }
        const f4 bb0 = *(const f4*)(biasrow + 16 * mt + 8 * sub);
        const f4 bb1 = *(const f4*)(biasrow + 16 * mt + 8 * sub + 4);
        v8f acc;
#pragma unroll
        for (int r = 0; r < 4; ++r) { acc[r] = bb0[r]; acc[4 + r] = bb1[r]; }
#pragma unroll
        for (int kt = 0; kt < 4; ++kt)
          acc = __builtin_amdgcn_wmma_f32_16x16x32_f16(
                    false, wa[kt], false, b0[kt], (short)0, acc, false, false);
#pragma unroll
        for (int r = 0; r < 8; ++r)
          r1[mt * 8 + r] = __sinf(W0f * acc[r]);   // D layout == h layout
      }
    }

    // B-fragments of r1 for layer 1 (register-only, lane-local)
    v16h b1[4];
    build_b(r1, subsel, b1);

    // ====== layer 1: r2 = sin(W1t x r1^T + b); h = 0.5*(h + r2) =============
    {
      const _Float16* wl      = &smem_w[1][0];
      const float*    biasrow = bblk + (blk * NLAY + 1) * HID;
#pragma unroll
      for (int mt = 0; mt < 8; ++mt) {
        const _Float16* wr = wl + (16 * mt + lm) * HID + 8 * sub;
        v16h wa[4];
#pragma unroll
        for (int kt = 0; kt < 4; ++kt) {
          const v8h lo = *(const v8h*)(wr + 32 * kt);
          const v8h hi = *(const v8h*)(wr + 32 * kt + 16);
#pragma unroll
          for (int i = 0; i < 8; ++i) { wa[kt][i] = lo[i]; wa[kt][8 + i] = hi[i]; }
        }
        const f4 bb0 = *(const f4*)(biasrow + 16 * mt + 8 * sub);
        const f4 bb1 = *(const f4*)(biasrow + 16 * mt + 8 * sub + 4);
        v8f acc;
#pragma unroll
        for (int r = 0; r < 4; ++r) { acc[r] = bb0[r]; acc[4 + r] = bb1[r]; }
#pragma unroll
        for (int kt = 0; kt < 4; ++kt)
          acc = __builtin_amdgcn_wmma_f32_16x16x32_f16(
                    false, wa[kt], false, b1[kt], (short)0, acc, false, false);
#pragma unroll
        for (int r = 0; r < 8; ++r)
          h[mt * 8 + r] = 0.5f * (h[mt * 8 + r] + __sinf(W0f * acc[r]));
      }
    }
  }

  // ---- write out [N,128], coalesced 128B/row segments ----
  float* orow = out + row * (long)HID;
#pragma unroll
  for (int g = 0; g < 8; ++g) {
    const int c0 = 16 * g + 8 * sub;
    f4 o0, o1;
#pragma unroll
    for (int e = 0; e < 4; ++e) { o0[e] = h[g * 8 + e]; o1[e] = h[g * 8 + 4 + e]; }
    *(f4*)(orow + c0)     = o0;
    *(f4*)(orow + c0 + 4) = o1;
  }
}

// ---------------------------------------------------------------------------
extern "C" void kernel_launch(void* const* d_in, const int* in_sizes, int n_in,
                              void* d_out, int out_size, void* d_ws, size_t ws_size,
                              hipStream_t stream) {
  const float* x       = (const float*)d_in[0];
  const float* gammas  = (const float*)d_in[1];
  const float* betas   = (const float*)d_in[2];
  const float* Wpre    = (const float*)d_in[3];
  const float* bpre    = (const float*)d_in[4];
  const float* Wblocks = (const float*)d_in[5];
  const float* bblocks = (const float*)d_in[6];
  float*       out     = (float*)d_out;
  _Float16*    Wt      = (_Float16*)d_ws;   // needs 16*128*128*2 = 512 KB

  const int N = in_sizes[0] / 4;            // x is [N,4]

  // pre-pass: f32 -> transposed f16 weights (deterministic every call)
  wt_f16_kernel<<<(NBLK * NLAY * HID * HID) / 256, 256, 0, stream>>>(Wblocks, Wt);

  // main: 8 waves/block, 16 rows/wave -> 128 rows/block
  siren_film_wmma<<<N / 128, 256, 0, stream>>>(x, gammas, betas, Wpre, bpre,
                                               Wt, bblocks, out);
}
